// Transform_Layer_44633300140607
// MI455X (gfx1250) — matrix operand order repaired and checked by reading, not compiled, not run on hardware
//
#include <hip/hip_runtime.h>
#include <hip/hip_bf16.h>

typedef __attribute__((ext_vector_type(16))) __bf16 v16bf;
typedef __attribute__((ext_vector_type(8)))  float  v8f;

#define PL     (1024u * 1024u)
#define NFREQ  17
#define TW32   0.19634954084936207f   // 2*pi/32

__device__ __forceinline__ unsigned short f2bf(float f) {
    unsigned int u = __float_as_uint(f);
    u += 0x7FFFu + ((u >> 16) & 1u);   // round-to-nearest-even
    return (unsigned short)(u >> 16);
}

// ---- CDNA5 async global->LDS copy (ASYNCcnt path), with sync fallback -----
#if __has_builtin(__builtin_amdgcn_global_load_async_to_lds_b128)
#define HAVE_ASYNC_LDS 1
__device__ __forceinline__ void async_cp16(const unsigned short* g, unsigned short* l) {
    typedef int v4i __attribute__((vector_size(4 * sizeof(int))));
    typedef __attribute__((address_space(1))) v4i gv4i;   // global
    typedef __attribute__((address_space(3))) v4i lv4i;   // LDS
    __builtin_amdgcn_global_load_async_to_lds_b128((gv4i*)g, (lv4i*)l, 0, 0);
}
#else
#define HAVE_ASYNC_LDS 0
__device__ __forceinline__ void async_cp16(const unsigned short* g, unsigned short* l) {
    *(uint4*)l = *(const uint4*)g;   // fallback: through VGPRs
}
#endif

#if __has_builtin(__builtin_amdgcn_s_wait_asynccnt)
#define WAIT_ASYNC(n) __builtin_amdgcn_s_wait_asynccnt(n)
#else
#define WAIT_ASYNC(n) asm volatile("s_wait_asynccnt %0" ::"n"(n) : "memory")
#endif

// ---------------------------------------------------------------------------
// Kernel 1: forward DFT (length 32, real input) of weights along tube axis.
// weights: [32][1024(p)][1024(q)] f32 -> Wf[k][p][q] bf16, k = 0..16.
// ---------------------------------------------------------------------------
__global__ void __launch_bounds__(256) dft_w_kernel(const float* __restrict__ w,
                                                    unsigned short* __restrict__ wre,
                                                    unsigned short* __restrict__ wim) {
    const unsigned idx = blockIdx.x * 256u + threadIdx.x;   // flat (p,q)
    float ct[32], st[32];
#pragma unroll
    for (int t = 0; t < 32; ++t) __sincosf((float)t * TW32, &st[t], &ct[t]);
    float xv[32];
#pragma unroll
    for (int j = 0; j < 32; ++j) xv[j] = w[idx + (unsigned)j * PL];
#pragma unroll
    for (int k = 0; k < NFREQ; ++k) {
        float re = 0.f, im = 0.f;
#pragma unroll
        for (int j = 0; j < 32; ++j) {
            const int t = (k * j) & 31;
            re = fmaf(xv[j], ct[t], re);
            im = fmaf(xv[j], -st[t], im);
        }
        wre[(unsigned)k * PL + idx] = f2bf(re);
        wim[(unsigned)k * PL + idx] = f2bf(im);
    }
}

// ---------------------------------------------------------------------------
// Kernel 2: forward DFT of x along tube axis, stored TRANSPOSED per plane:
// x: [32][1024(q)][1024(m)] f32 -> XfT[k][m][q] bf16 (so WMMA B-fragments
// read contiguous K). 32x32 tile transpose through LDS. 1024 threads/block.
// ---------------------------------------------------------------------------
__global__ void __launch_bounds__(1024) dft_x_kernel(const float* __restrict__ x,
                                                     unsigned short* __restrict__ xre,
                                                     unsigned short* __restrict__ xim) {
    __shared__ float sRe[32][33];
    __shared__ float sIm[32][33];
    const int tx = threadIdx.x & 31;    // m within tile
    const int ty = threadIdx.x >> 5;    // q within tile
    const unsigned m0 = blockIdx.x * 32u;
    const unsigned q0 = blockIdx.y * 32u;

    float ct[32], st[32];
#pragma unroll
    for (int t = 0; t < 32; ++t) __sincosf((float)t * TW32, &st[t], &ct[t]);

    float xv[32];
    const unsigned base = (q0 + (unsigned)ty) * 1024u + (m0 + (unsigned)tx);
#pragma unroll
    for (int j = 0; j < 32; ++j) xv[j] = x[base + (unsigned)j * PL];

#pragma unroll
    for (int k = 0; k < NFREQ; ++k) {
        float re = 0.f, im = 0.f;
#pragma unroll
        for (int j = 0; j < 32; ++j) {
            const int t = (k * j) & 31;
            re = fmaf(xv[j], ct[t], re);
            im = fmaf(xv[j], -st[t], im);
        }
        sRe[ty][tx] = re;
        sIm[ty][tx] = im;
        __syncthreads();
        // transposed, coalesced along q
        const unsigned o = (unsigned)k * PL + (m0 + (unsigned)ty) * 1024u + (q0 + (unsigned)tx);
        xre[o] = f2bf(sRe[tx][ty]);
        xim[o] = f2bf(sIm[tx][ty]);
        __syncthreads();
    }
}

// ---------------------------------------------------------------------------
// Kernel 3: batched complex GEMM per frequency, bf16 WMMA, f32 accumulate.
// Cf[k] = Wf[k] @ Xf[k].  Block: 512 thr (16 waves), 128x128 tile, Kc=32,
// DOUBLE-BUFFERED LDS fed by GLOBAL_LOAD_ASYNC_TO_LDS_B128 (ASYNCcnt).
// Wave tile 32x32 = 2x2 WMMA tiles. bf16 WMMA has no A/B negate, so keep
// three accumulators and do Cre = rr - ii in the epilogue.
// ---------------------------------------------------------------------------
#define LDSP 40   // padded LDS row stride (ushorts): 80B -> conflict-free b128 frag reads
#define NCHUNK 32 // 1024 / 32

__global__ void __launch_bounds__(512) cgemm_kernel(const unsigned short* __restrict__ wre,
                                                    const unsigned short* __restrict__ wim,
                                                    const unsigned short* __restrict__ xre,
                                                    const unsigned short* __restrict__ xim,
                                                    float* __restrict__ cre,
                                                    float* __restrict__ cim) {
    __shared__ __align__(16) unsigned short sAre[2][128][LDSP];
    __shared__ __align__(16) unsigned short sAim[2][128][LDSP];
    __shared__ __align__(16) unsigned short sBre[2][128][LDSP];
    __shared__ __align__(16) unsigned short sBim[2][128][LDSP];

    const unsigned kf = blockIdx.z;
    const unsigned short* Wre = wre + (size_t)kf * PL;
    const unsigned short* Wim = wim + (size_t)kf * PL;
    const unsigned short* Xre = xre + (size_t)kf * PL;
    const unsigned short* Xim = xim + (size_t)kf * PL;
    float* Cre = cre + (size_t)kf * PL;
    float* Cim = cim + (size_t)kf * PL;

    const unsigned tid    = threadIdx.x;
    const unsigned lane   = tid & 31u;
    const unsigned wv     = tid >> 5;        // 0..15
    const unsigned waveM  = wv >> 2;         // 0..3 (p)
    const unsigned waveN  = wv & 3u;         // 0..3 (m)
    const unsigned laneHi = lane >> 4;       // 0/1
    const unsigned laneLo = lane & 15u;

    const unsigned pBlk = blockIdx.y * 128u;
    const unsigned mBlk = blockIdx.x * 128u;

    const unsigned ldRow = tid >> 2;         // 0..127
    const unsigned ldCol = (tid & 3u) * 8u;  // 0,8,16,24 (ushorts)

    const v8f vzero = {0.f, 0.f, 0.f, 0.f, 0.f, 0.f, 0.f, 0.f};
    v8f acc_rr[2][2], acc_ii[2][2], acc_im[2][2];
#pragma unroll
    for (int i = 0; i < 2; ++i)
#pragma unroll
        for (int j = 0; j < 2; ++j) {
            acc_rr[i][j] = vzero; acc_ii[i][j] = vzero; acc_im[i][j] = vzero;
        }

    // stage one 128x32 K-chunk (4 tiles, 1 async b128 per thread per tile)
    auto stage = [&](unsigned buf, unsigned kk) {
        const unsigned aOff = (pBlk + ldRow) * 1024u + kk + ldCol;
        const unsigned bOff = (mBlk + ldRow) * 1024u + kk + ldCol;
        async_cp16(Wre + aOff, &sAre[buf][ldRow][ldCol]);
        async_cp16(Wim + aOff, &sAim[buf][ldRow][ldCol]);
        async_cp16(Xre + bOff, &sBre[buf][ldRow][ldCol]);
        async_cp16(Xim + bOff, &sBim[buf][ldRow][ldCol]);
    };

    stage(0u, 0u);   // prefetch chunk 0

    for (unsigned ci = 0; ci < NCHUNK; ++ci) {
        const unsigned cur = ci & 1u;
        if (ci + 1u < NCHUNK) {
            stage(cur ^ 1u, (ci + 1u) * 32u);  // overlap: next chunk in flight
            WAIT_ASYNC(4);                     // in-order completion => chunk ci landed
        } else {
            WAIT_ASYNC(0);
        }
        __syncthreads();                       // all waves' async data visible

        v16bf aRe[2], aIm[2], bRe[2], bIm[2];
#pragma unroll
        for (int mt = 0; mt < 2; ++mt) {
            // A 16x32 bf16: lanes 0-15 K=0..7|16..23, lanes 16-31 K=8..15|24..31
            const unsigned r  = waveM * 32u + (unsigned)mt * 16u + laneLo;
            const unsigned c0 = 8u * laneHi;
            const unsigned c1 = 16u + 8u * laneHi;
            ((uint4*)&aRe[mt])[0] = *(const uint4*)&sAre[cur][r][c0];
            ((uint4*)&aRe[mt])[1] = *(const uint4*)&sAre[cur][r][c1];
            ((uint4*)&aIm[mt])[0] = *(const uint4*)&sAim[cur][r][c0];
            ((uint4*)&aIm[mt])[1] = *(const uint4*)&sAim[cur][r][c1];
        }
#pragma unroll
        for (int nt = 0; nt < 2; ++nt) {
            // B 32x16 bf16: lane = column (m); lanes 0-15 K=0..7|16..23, 16-31 K=8..15|24..31
            const unsigned r  = waveN * 32u + (unsigned)nt * 16u + laneLo;
            const unsigned c0 = 8u * laneHi;
            ((uint4*)&bRe[nt])[0] = *(const uint4*)&sBre[cur][r][c0];
            ((uint4*)&bRe[nt])[1] = *(const uint4*)&sBre[cur][r][c0 + 16u];
            ((uint4*)&bIm[nt])[0] = *(const uint4*)&sBim[cur][r][c0];
            ((uint4*)&bIm[nt])[1] = *(const uint4*)&sBim[cur][r][c0 + 16u];
        }
#pragma unroll
        for (int mt = 0; mt < 2; ++mt)
#pragma unroll
            for (int nt = 0; nt < 2; ++nt) {
                acc_rr[mt][nt] = __builtin_amdgcn_wmma_f32_16x16x32_bf16(
                    false, aRe[mt], false, bRe[nt], (short)0, acc_rr[mt][nt], false, false);
                acc_ii[mt][nt] = __builtin_amdgcn_wmma_f32_16x16x32_bf16(
                    false, aIm[mt], false, bIm[nt], (short)0, acc_ii[mt][nt], false, false);
                acc_im[mt][nt] = __builtin_amdgcn_wmma_f32_16x16x32_bf16(
                    false, aRe[mt], false, bIm[nt], (short)0, acc_im[mt][nt], false, false);
                acc_im[mt][nt] = __builtin_amdgcn_wmma_f32_16x16x32_bf16(
                    false, aIm[mt], false, bRe[nt], (short)0, acc_im[mt][nt], false, false);
            }
        __syncthreads();   // all reads of buf `cur` done before it is re-filled
    }

    // Epilogue: C/D layout = VGPR v: lanes 0-15 -> (M=v, N=lane); 16-31 -> (M=v+8)
#pragma unroll
    for (int mt = 0; mt < 2; ++mt)
#pragma unroll
        for (int nt = 0; nt < 2; ++nt)
#pragma unroll
            for (int v = 0; v < 8; ++v) {
                const unsigned p = pBlk + waveM * 32u + (unsigned)mt * 16u + (unsigned)v + 8u * laneHi;
                const unsigned m = mBlk + waveN * 32u + (unsigned)nt * 16u + laneLo;
                Cre[p * 1024u + m] = acc_rr[mt][nt][v] - acc_ii[mt][nt][v];
                Cim[p * 1024u + m] = acc_im[mt][nt][v];
            }
}

// ---------------------------------------------------------------------------
// Kernel 4: inverse DFT (conjugate symmetry) + bias.
// out[n][p][m] = (1/32)(re0 + (-1)^n re16 + 2*sum_{k=1..15}(re_k cos - im_k sin)) + bias[p][m]
// ---------------------------------------------------------------------------
__global__ void __launch_bounds__(256) ifft_bias_kernel(const float* __restrict__ cre,
                                                        const float* __restrict__ cim,
                                                        const float* __restrict__ bias,
                                                        float* __restrict__ out) {
    const unsigned idx = blockIdx.x * 256u + threadIdx.x;  // flat (p,m)
    float ct[32], st[32];
#pragma unroll
    for (int t = 0; t < 32; ++t) __sincosf((float)t * TW32, &st[t], &ct[t]);

    float re[NFREQ], im[NFREQ];
#pragma unroll
    for (int k = 0; k < NFREQ; ++k) {
        re[k] = cre[(unsigned)k * PL + idx];
        im[k] = cim[(unsigned)k * PL + idx];
    }
#pragma unroll
    for (int k = 1; k < 16; ++k) { re[k] *= 2.f; im[k] *= 2.f; }
    const float b = bias[idx];

#pragma unroll
    for (int n = 0; n < 32; ++n) {
        float acc = re[0] + ((n & 1) ? -re[16] : re[16]);
#pragma unroll
        for (int k = 1; k < 16; ++k) {
            const int t = (k * n) & 31;
            acc = fmaf(re[k], ct[t], acc);
            acc = fmaf(-im[k], st[t], acc);
        }
        out[(unsigned)n * PL + idx] = acc * (1.f / 32.f) + b;
    }
}

// ---------------------------------------------------------------------------
extern "C" void kernel_launch(void* const* d_in, const int* in_sizes, int n_in,
                              void* d_out, int out_size, void* d_ws, size_t ws_size,
                              hipStream_t stream) {
    const float* x    = (const float*)d_in[0];   // [32][1024][1024]
    const float* w    = (const float*)d_in[1];   // [32][1024][1024]
    const float* bias = (const float*)d_in[2];   // [1][1024][1024]
    float* out = (float*)d_out;                  // [32][1024][1024]

    // Workspace layout (~285 MB):
    //   Wf re/im, XfT re/im: 4 * 17 planes bf16; Cf re/im: 2 * 17 planes f32.
    unsigned short* Wre = (unsigned short*)d_ws;
    unsigned short* Wim = Wre + (size_t)NFREQ * PL;
    unsigned short* Xre = Wim + (size_t)NFREQ * PL;
    unsigned short* Xim = Xre + (size_t)NFREQ * PL;
    float* Cre = (float*)(Xim + (size_t)NFREQ * PL);
    float* Cim = Cre + (size_t)NFREQ * PL;

    dft_w_kernel<<<4096, 256, 0, stream>>>(w, Wre, Wim);
    dft_x_kernel<<<dim3(32, 32), 1024, 0, stream>>>(x, Xre, Xim);
    cgemm_kernel<<<dim3(8, 8, NFREQ), 512, 0, stream>>>(Wre, Wim, Xre, Xim, Cre, Cim);
    ifft_bias_kernel<<<4096, 256, 0, stream>>>(Cre, Cim, bias, out);
}
